// GTAM_21852793602070
// MI455X (gfx1250) — matrix operand order repaired and checked by reading, not compiled, not run on hardware
//
#include <hip/hip_runtime.h>
#include <hip/hip_bf16.h>

#define BATCH 2
#define NCH   96
#define NTOK  6400   // 80*80
#define QKVC  288    // 3*96

typedef __attribute__((ext_vector_type(16))) __bf16 v16bf;
typedef __attribute__((ext_vector_type(8)))  __bf16 v8bf;
typedef __attribute__((ext_vector_type(8)))  float  v8f;

// ---------------------------------------------------------------------------
// WMMA helpers: layouts per CDNA5 ISA 7.12.2 (wave32)
// A (16x32 bf16): lane L holds row M=L&15; K = kbase + {0..7, 16..23}, kbase=(L>>4)*8
// B (32x16 bf16): lane L holds col N=L&15; K = kbase + {0..7, 16..23}, kbase=(L>>4)*8
// C/D (16x16 f32): lane L holds col N=L&15; VGPR r = row (L>>4)*8 + r
// ---------------------------------------------------------------------------
__device__ __forceinline__ v8f wmma_bf16(v16bf a, v16bf b, v8f c) {
  return __builtin_amdgcn_wmma_f32_16x16x32_bf16(false, a, false, b, (short)0, c,
                                                 false, false);
}

// A tile from row-major (M x K) storage, src -> tile origin, ld in elements
__device__ __forceinline__ v16bf load_A(const __bf16* src, int ld, int lane) {
  int m = lane & 15;
  int kbase = (lane >> 4) * 8;
  const __bf16* p = src + m * ld + kbase;
  v8bf lo = *(const v8bf*)(p);
  v8bf hi = *(const v8bf*)(p + 16);
  v16bf a;
#pragma unroll
  for (int e = 0; e < 8; ++e) { a[e] = lo[e]; a[e + 8] = hi[e]; }
  return a;
}

// B tile where memory holds (N x K) row-major (i.e. B = memᵀ): B[k][n] = src[n*ld + k]
// Per lane K = kbase + {0..7, 16..23} (same order as load_A) -> two 16B vector loads.
__device__ __forceinline__ v16bf load_B_nk(const __bf16* src, int ld, int lane) {
  int n = lane & 15;
  int kbase = (lane >> 4) * 8;
  const __bf16* p = src + n * ld + kbase;
  v8bf lo = *(const v8bf*)(p);
  v8bf hi = *(const v8bf*)(p + 16);
  v16bf b;
#pragma unroll
  for (int e = 0; e < 8; ++e) { b[e] = lo[e]; b[e + 8] = hi[e]; }
  return b;
}

// Store 16x16 f32 C tile to row-major dst
__device__ __forceinline__ void store_C(float* dst, int ld, int lane, v8f c) {
  int n = lane & 15;
  int mbase = (lane >> 4) * 8;
#pragma unroll
  for (int r = 0; r < 8; ++r) dst[(mbase + r) * ld + n] = c[r];
}

// ---------------------------------------------------------------------------
// K1: conv1x1 qkv, both branches.  y1[(br*B+b)*288 + o][n] f32
// ---------------------------------------------------------------------------
__global__ void k_conv1x1(const float* __restrict__ x,
                          const float* __restrict__ wc, const float* __restrict__ bc,
                          const float* __restrict__ wp, const float* __restrict__ bp,
                          float* __restrict__ y1) {
  int blk = blockIdx.x;
  int o  = blk % QKVC;
  int b  = (blk / QKVC) % BATCH;
  int br = blk / (QKVC * BATCH);
  const float* w = (br == 0 ? wc : wp) + o * NCH;
  float bias = (br == 0 ? bc : bp)[o];
  __shared__ float sw[NCH];
  if (threadIdx.x < NCH) sw[threadIdx.x] = w[threadIdx.x];
  __syncthreads();
  const float* xb = x + b * NCH * NTOK;
  float* dst = y1 + (size_t)((br * BATCH + b) * QKVC + o) * NTOK;
  for (int n = threadIdx.x; n < NTOK; n += blockDim.x) {
    float acc = bias;
#pragma unroll 8
    for (int c = 0; c < NCH; ++c) acc += sw[c] * xb[c * NTOK + n];
    dst[n] = acc;
  }
}

// ---------------------------------------------------------------------------
// K2: depthwise 3x3 + bias, split q/k/v, cast to bf16.
// Layouts chosen so every later WMMA B-operand is an (N x K) contiguous load:
//   CTA: q,k channel-major (b,c,n);  v token-major (b,n,c)
//   PTA: q,k token-major  (b,n,c);   v channel-major (b,c,n)
// ---------------------------------------------------------------------------
__global__ void k_dwconv_split(const float* __restrict__ y1,
                               const float* __restrict__ dwc, const float* __restrict__ dbc,
                               const float* __restrict__ dwp, const float* __restrict__ dbp,
                               __bf16* __restrict__ cq, __bf16* __restrict__ ck,
                               __bf16* __restrict__ cv,
                               __bf16* __restrict__ pq, __bf16* __restrict__ pk,
                               __bf16* __restrict__ pv) {
  int blk = blockIdx.x;
  int o  = blk % QKVC;
  int b  = (blk / QKVC) % BATCH;
  int br = blk / (QKVC * BATCH);
  const float* w9 = (br == 0 ? dwc : dwp) + o * 9;
  float bias = (br == 0 ? dbc : dbp)[o];
  float w[9];
#pragma unroll
  for (int t = 0; t < 9; ++t) w[t] = w9[t];
  const float* src = y1 + (size_t)((br * BATCH + b) * QKVC + o) * NTOK;
  int part = o / NCH, c = o % NCH;
  __bf16* dq = (br == 0) ? (part == 0 ? cq : part == 1 ? ck : cv)
                         : (part == 0 ? pq : part == 1 ? pk : pv);
  bool chanMajor = (br == 0) ? (part < 2) : (part == 2);
  for (int n = threadIdx.x; n < NTOK; n += blockDim.x) {
    int hh = n / 80, ww = n % 80;
    float acc = bias;
#pragma unroll
    for (int dy = -1; dy <= 1; ++dy) {
#pragma unroll
      for (int dx = -1; dx <= 1; ++dx) {
        int yy = hh + dy, xx = ww + dx;
        if (yy >= 0 && yy < 80 && xx >= 0 && xx < 80)
          acc += w[(dy + 1) * 3 + (dx + 1)] * src[yy * 80 + xx];
      }
    }
    __bf16 val = (__bf16)acc;
    if (chanMajor) dq[((size_t)b * NCH + c) * NTOK + n] = val;
    else           dq[((size_t)b * NTOK + n) * NCH + c] = val;
  }
}

// ---------------------------------------------------------------------------
// K3a: CTA dots[c][d] = sum_n q[c,n] k[d,n]   (96x96, K=6400), WMMA bf16
// ---------------------------------------------------------------------------
__global__ void k_cta_dots(const __bf16* __restrict__ q, const __bf16* __restrict__ k,
                           float* __restrict__ dots) {
  int tile = blockIdx.x;              // 36 tiles of 16x16
  int b = blockIdx.y;
  int ci = tile / 6, di = tile % 6;
  int lane = threadIdx.x & 31;
  const __bf16* qb = q + ((size_t)b * NCH + ci * 16) * NTOK;
  const __bf16* kb = k + ((size_t)b * NCH + di * 16) * NTOK;
  v8f acc; for (int r = 0; r < 8; ++r) acc[r] = 0.f;
  for (int kk = 0; kk < NTOK; kk += 32)
    acc = wmma_bf16(load_A(qb + kk, NTOK, lane), load_B_nk(kb + kk, NTOK, lane), acc);
  store_C(dots + ((size_t)b * NCH + ci * 16) * NCH + di * 16, NCH, lane, acc);
}

// ---------------------------------------------------------------------------
// K3b: row softmax of 96-wide rows, write bf16 attn
// ---------------------------------------------------------------------------
__global__ void k_cta_softmax(const float* __restrict__ dots, __bf16* __restrict__ attn) {
  int r = blockIdx.x, b = blockIdx.y, t = threadIdx.x;
  const float* row = dots + ((size_t)b * NCH + r) * NCH;
  __shared__ float sh[NCH];
  __shared__ float smax, ssum;
  if (t < NCH) sh[t] = row[t];
  __syncthreads();
  if (t == 0) { float m = -1e30f; for (int i = 0; i < NCH; ++i) m = fmaxf(m, sh[i]); smax = m; }
  __syncthreads();
  if (t < NCH) sh[t] = __expf(sh[t] - smax);
  __syncthreads();
  if (t == 0) { float s = 0.f; for (int i = 0; i < NCH; ++i) s += sh[i]; ssum = s; }
  __syncthreads();
  if (t < NCH) attn[((size_t)b * NCH + r) * NCH + t] = (__bf16)(sh[t] / ssum);
}

// ---------------------------------------------------------------------------
// K3c: out_cta[c][n] = sum_d attn[c][d] v[d][n]  (96 x 6400, K=96), WMMA.
// cv is token-major (n,c): B[d][n] = cv[n][d] -> load_B_nk, contiguous.
// ---------------------------------------------------------------------------
__global__ void k_cta_out(const __bf16* __restrict__ attn, const __bf16* __restrict__ cv,
                          float* __restrict__ outc) {
  int wave = threadIdx.x >> 5, lane = threadIdx.x & 31;
  int tile = blockIdx.x * 8 + wave;   // 2400 tiles: 6 x 400
  int b = blockIdx.y;
  int ci = tile / 400, ni = tile % 400;
  const __bf16* ab = attn + ((size_t)b * NCH + ci * 16) * NCH;
  v8f acc; for (int r = 0; r < 8; ++r) acc[r] = 0.f;
#pragma unroll
  for (int kk = 0; kk < NCH; kk += 32)
    acc = wmma_bf16(load_A(ab + kk, NCH, lane),
                    load_B_nk(cv + ((size_t)b * NTOK + ni * 16) * NCH + kk, NCH, lane),
                    acc);
  store_C(outc + ((size_t)b * NCH + ci * 16) * NTOK + ni * 16, NTOK, lane, acc);
}

// ---------------------------------------------------------------------------
// K4: PTA flash attention.  Block = 8 waves = 128 query rows.
// K-tile (32x96) and V-tile (96x32) staged in LDS once per block and shared
// by all 8 waves (8x cut in L2 traffic); all WMMA B-operands come from LDS.
// q,k token-major (b,n,c);  v channel-major (b,c,n).  Out (b,n,c) f32.
// ---------------------------------------------------------------------------
__global__ void k_pta_attn(const __bf16* __restrict__ q, const __bf16* __restrict__ k,
                           const __bf16* __restrict__ v, float* __restrict__ outp) {
  int tid = threadIdx.x;
  int wave = tid >> 5, lane = tid & 31;
  int b = blockIdx.y;
  int q0 = (blockIdx.x * 8 + wave) * 16;

  __shared__ __align__(16) __bf16 Ktile[32 * NCH];   // [key][ch]   6 KB
  __shared__ __align__(16) __bf16 Vtile[NCH * 32];   // [ch][key]   6 KB
  __shared__ __align__(16) __bf16 plds[8][16 * 32];  // per-wave P  8 KB

  const __bf16* qb = q + ((size_t)b * NTOK + q0) * NCH;
  v16bf aQ[3];
#pragma unroll
  for (int t = 0; t < 3; ++t) aQ[t] = load_A(qb + 32 * t, NCH, lane);

  v8f O[6];
#pragma unroll
  for (int ch = 0; ch < 6; ++ch)
    for (int r = 0; r < 8; ++r) O[ch][r] = 0.f;
  float rmax[8], rsum[8];
#pragma unroll
  for (int r = 0; r < 8; ++r) { rmax[r] = -1e30f; rsum[r] = 0.f; }
  int mbase = (lane >> 4) * 8;
  int ncol = lane & 15;

  for (int jb = 0; jb < NTOK; jb += 32) {
    // --- cooperative stage of K/V tiles into LDS (16B chunks) ---
    const __bf16* kg = k + ((size_t)b * NTOK + jb) * NCH;          // 32 rows x 192B
    const __bf16* vg = v + (size_t)b * NCH * NTOK + jb;            // 96 rows x 64B
    for (int idx = tid; idx < 384; idx += 256) {
      int kr = idx / 12, kc = (idx % 12) * 8;                      // K: [32][96]
      *(v8bf*)(Ktile + kr * NCH + kc) = *(const v8bf*)(kg + kr * NCH + kc);
      int vr = idx / 4, vc = (idx % 4) * 8;                        // V: [96][32]
      *(v8bf*)(Vtile + vr * 32 + vc) = *(const v8bf*)(vg + (size_t)vr * NTOK + vc);
    }
    // prefetch next tiles into cache while we compute on this one
    int jn = (jb + 32 < NTOK) ? jb + 32 : jb;
    __builtin_prefetch(k + ((size_t)b * NTOK + jn) * NCH + (size_t)(tid % 96) * 64, 0, 1);
    __builtin_prefetch(v + ((size_t)b * NCH + (tid % 96)) * NTOK + jn, 0, 1);
    __syncthreads();

    // --- S = Q Ktileᵀ : two 16x16 tiles, K=96 ---
    v8f S0, S1;
    for (int r = 0; r < 8; ++r) { S0[r] = 0.f; S1[r] = 0.f; }
#pragma unroll
    for (int t = 0; t < 3; ++t) {
      S0 = wmma_bf16(aQ[t], load_B_nk(Ktile + 32 * t, NCH, lane), S0);
      S1 = wmma_bf16(aQ[t], load_B_nk(Ktile + 16 * NCH + 32 * t, NCH, lane), S1);
    }
    // --- online softmax: row = mbase+r lives in the 16 lanes with same lane>>4 ---
    float fac[8];
#pragma unroll
    for (int r = 0; r < 8; ++r) {
      float m = fmaxf(S0[r], S1[r]);
      for (int d = 1; d < 16; d <<= 1) m = fmaxf(m, __shfl_xor(m, d, 32));
      float nm = fmaxf(rmax[r], m);
      fac[r] = __expf(rmax[r] - nm);
      rmax[r] = nm;
    }
    float p0[8], p1[8];
#pragma unroll
    for (int r = 0; r < 8; ++r) {
      p0[r] = __expf(S0[r] - rmax[r]);
      p1[r] = __expf(S1[r] - rmax[r]);
      float ls = p0[r] + p1[r];
      for (int d = 1; d < 16; d <<= 1) ls += __shfl_xor(ls, d, 32);
      rsum[r] = rsum[r] * fac[r] + ls;
    }
    // --- C-layout -> A-layout via per-wave LDS round trip ---
    __bf16* pw = plds[wave];
#pragma unroll
    for (int r = 0; r < 8; ++r) {
      pw[(mbase + r) * 32 + ncol]      = (__bf16)p0[r];
      pw[(mbase + r) * 32 + 16 + ncol] = (__bf16)p1[r];
    }
    v16bf aP = load_A(pw, 32, lane);
    // --- O += P Vtile : 6 channel chunks, K=32 ---
#pragma unroll
    for (int ch = 0; ch < 6; ++ch) {
#pragma unroll
      for (int r = 0; r < 8; ++r) O[ch][r] *= fac[r];
      O[ch] = wmma_bf16(aP, load_B_nk(Vtile + ch * 16 * 32, 32, lane), O[ch]);
    }
    __syncthreads();   // before next iteration overwrites K/V tiles
  }
  float* ob = outp + ((size_t)b * NTOK + q0) * NCH;
#pragma unroll
  for (int ch = 0; ch < 6; ++ch) {
    v8f t = O[ch];
#pragma unroll
    for (int r = 0; r < 8; ++r) t[r] /= rsum[r];
    store_C(ob + ch * 16, NCH, lane, t);
  }
}

// ---------------------------------------------------------------------------
// K5: both proj conv1x1 + combine 0.01*cta + pta, output (b, n, c) f32
// ---------------------------------------------------------------------------
__global__ void k_proj_combine(const float* __restrict__ outc, const float* __restrict__ outp,
                               const float* __restrict__ wc, const float* __restrict__ bc,
                               const float* __restrict__ wp, const float* __restrict__ bp,
                               float* __restrict__ out) {
  int idx = blockIdx.x * blockDim.x + threadIdx.x;
  if (idx >= BATCH * NTOK * NCH) return;
  int co = idx % NCH;
  int n  = (idx / NCH) % NTOK;
  int b  = idx / (NCH * NTOK);
  const float* wcr = wc + co * NCH;
  const float* wpr = wp + co * NCH;
  const float* cb = outc + (size_t)b * NCH * NTOK + n;       // (b,c,n), stride NTOK
  const float* pb = outp + ((size_t)b * NTOK + n) * NCH;     // (b,n,c), contiguous
  float accC = bc[co], accP = bp[co];
#pragma unroll 8
  for (int c = 0; c < NCH; ++c) {
    accC += wcr[c] * cb[(size_t)c * NTOK];
    accP += wpr[c] * pb[c];
  }
  out[idx] = accC * 0.01f + accP;
}

// ---------------------------------------------------------------------------
extern "C" void kernel_launch(void* const* d_in, const int* in_sizes, int n_in,
                              void* d_out, int out_size, void* d_ws, size_t ws_size,
                              hipStream_t stream) {
  const float* x         = (const float*)d_in[0];
  const float* cta_qkv_w = (const float*)d_in[1];
  const float* cta_qkv_b = (const float*)d_in[2];
  const float* cta_dw_w  = (const float*)d_in[3];
  const float* cta_dw_b  = (const float*)d_in[4];
  const float* cta_pj_w  = (const float*)d_in[5];
  const float* cta_pj_b  = (const float*)d_in[6];
  const float* pta_qkv_w = (const float*)d_in[7];
  const float* pta_qkv_b = (const float*)d_in[8];
  const float* pta_dw_w  = (const float*)d_in[9];
  const float* pta_dw_b  = (const float*)d_in[10];
  const float* pta_pj_w  = (const float*)d_in[11];
  const float* pta_pj_b  = (const float*)d_in[12];

  size_t off = 0;
  auto take = [&](size_t bytes) -> void* {
    void* p = (char*)d_ws + off;
    off += (bytes + 255) & ~(size_t)255;
    return p;
  };
  const size_t nBCN = (size_t)BATCH * NCH * NTOK;
  float*  y1   = (float*)take((size_t)2 * BATCH * QKVC * NTOK * 4);
  __bf16* cq   = (__bf16*)take(nBCN * 2);
  __bf16* ck   = (__bf16*)take(nBCN * 2);
  __bf16* cv   = (__bf16*)take(nBCN * 2);
  __bf16* pq   = (__bf16*)take(nBCN * 2);
  __bf16* pk   = (__bf16*)take(nBCN * 2);
  __bf16* pv   = (__bf16*)take(nBCN * 2);
  float*  dots = (float*)take((size_t)BATCH * NCH * NCH * 4);
  __bf16* attn = (__bf16*)take((size_t)BATCH * NCH * NCH * 2);
  float*  outc = (float*)take(nBCN * 4);
  float*  outp = (float*)take(nBCN * 4);

  k_conv1x1<<<dim3(QKVC * BATCH * 2), 256, 0, stream>>>(
      x, cta_qkv_w, cta_qkv_b, pta_qkv_w, pta_qkv_b, y1);
  k_dwconv_split<<<dim3(QKVC * BATCH * 2), 256, 0, stream>>>(
      y1, cta_dw_w, cta_dw_b, pta_dw_w, pta_dw_b, cq, ck, cv, pq, pk, pv);
  k_cta_dots<<<dim3(36, BATCH), 32, 0, stream>>>(cq, ck, dots);
  k_cta_softmax<<<dim3(NCH, BATCH), 96, 0, stream>>>(dots, attn);
  k_cta_out<<<dim3(300, BATCH), 256, 0, stream>>>(attn, cv, outc);
  k_pta_attn<<<dim3(50, BATCH), 256, 0, stream>>>(pq, pk, pv, outp);
  int total = BATCH * NTOK * NCH;
  k_proj_combine<<<(total + 255) / 256, 256, 0, stream>>>(
      outc, outp, cta_pj_w, cta_pj_b, pta_pj_w, pta_pj_b, (float*)d_out);
}
